// DLRM_MULTI_IPU_61856118997705
// MI455X (gfx1250) — matrix operand order repaired and checked by reading, not compiled, not run on hardware
//
#include <hip/hip_runtime.h>
#include <hip/hip_bf16.h>
#include <math.h>

// ---------------- problem constants (from reference) ----------------
#define BB     4096      // batch
#define LL     64        // lookups per bag
#define NT     4         // tables
#define VV     500000    // rows per table
#define DD     128       // emb dim
#define DENSE  512

typedef __attribute__((ext_vector_type(16))) __bf16 v16bf;
typedef __attribute__((ext_vector_type(8)))  __bf16 v8bf;
typedef __attribute__((ext_vector_type(4)))  __bf16 v4bf;
typedef __attribute__((ext_vector_type(8)))  float  v8f;

union AFrag { v16bf v; v8bf h[2]; };

// ---------------------------------------------------------------
// fp32 -> bf16 elementwise convert (weights / dense input)
// ---------------------------------------------------------------
__global__ void cvt_f32_bf16(const float* __restrict__ in, __bf16* __restrict__ out, int n) {
    int i = blockIdx.x * blockDim.x + threadIdx.x;
    if (i < n) out[i] = (__bf16)in[i];
}

// ---------------------------------------------------------------
// Embedding-bag sum: one wave per (bag, table).
// Lane owns dims [lane*4, lane*4+4): float4 coalesced row loads,
// fp32 accumulate, bf16 store into feat[:, 128 + t*128 + d].
// Pure HBM-bandwidth kernel (~536 MB gathered per call).
// ---------------------------------------------------------------
__global__ void embed_pool(const float* __restrict__ tables,
                           const int*   __restrict__ xidx,
                           const int*   __restrict__ offsets,
                           __bf16*      __restrict__ feat) {
    int wave = (blockIdx.x * blockDim.x + threadIdx.x) >> 5;
    int lane = threadIdx.x & 31;
    int t = wave & (NT - 1);
    int b = wave >> 2;
    if (b >= BB) return;

    int start = offsets[b];
    int end   = (b == BB - 1) ? (BB * LL) : offsets[b + 1];

    const float* tab = tables + (size_t)t * VV * DD;
    int d = lane * 4;
    float4 acc = make_float4(0.f, 0.f, 0.f, 0.f);

#pragma unroll 4
    for (int r = start; r < end; ++r) {
        int idx = xidx[r * NT + t];
        const float4 v = *(const float4*)(tab + (size_t)idx * DD + d);
        acc.x += v.x; acc.y += v.y; acc.z += v.z; acc.w += v.w;
    }

    v4bf o;
    o.x = (__bf16)acc.x; o.y = (__bf16)acc.y;
    o.z = (__bf16)acc.z; o.w = (__bf16)acc.w;
    *(v4bf*)(feat + (size_t)b * 640 + 128 + t * 128 + d) = o;
}

// ---------------------------------------------------------------
// bf16 WMMA GEMM:  Out[M,N] = act( A[M,K] @ W[N,K]^T + bias[N] )
// One wave -> 32x32 output block (2x2 v_wmma_f32_16x16x32_bf16
// accumulators), K-loop step 32. Operands straight from L2
// (weights ~MBs, fully L2-resident on the 192 MB L2).
//
// A-fragment (16x32 bf16, ISA 7.12.2): lane m=lane&15, g=lane>>4
//   elems 0..7  = K kb+g*8..+7      (16 B contiguous)
//   elems 8..15 = K kb+16+g*8..+7   (16 B contiguous)
// B-fragment (32x16): lane n=lane&15 needs column n = row n of W:
//   elems 0..7 = W[n][kb+g*8..+7], elems 8..15 = W[n][kb+16+g*8..+7]
// C/D (16x16 f32): lane (m>=8?16:0)+n, VGPR r = m&7.
// ---------------------------------------------------------------
__global__ void gemm_bf16_wmma(const __bf16* __restrict__ A,
                               const __bf16* __restrict__ W,
                               const float*  __restrict__ bias,
                               __bf16*       __restrict__ Out,
                               int M, int N, int K, int ldo, int relu) {
    int wave = (blockIdx.x * blockDim.x + threadIdx.x) >> 5;
    int lane = threadIdx.x & 31;
    int nTiles = N >> 5;
    int totalTiles = (M >> 5) * nTiles;
    if (wave >= totalTiles) return;          // wave-uniform: EXEC stays all-ones

    int row0 = (wave / nTiles) << 5;
    int col0 = (wave % nTiles) << 5;
    int g   = lane >> 4;
    int l16 = lane & 15;

    v8f acc00 = {}, acc01 = {}, acc10 = {}, acc11 = {};

    const __bf16* pa = A + (size_t)(row0 + l16) * K + g * 8;
    const __bf16* pw = W + (size_t)(col0 + l16) * K + g * 8;
    const size_t rowsk16 = (size_t)16 * K;

    for (int kb = 0; kb < K; kb += 32) {
        AFrag a0, a1, b0, b1;
        a0.h[0] = *(const v8bf*)(pa + kb);
        a0.h[1] = *(const v8bf*)(pa + kb + 16);
        a1.h[0] = *(const v8bf*)(pa + rowsk16 + kb);
        a1.h[1] = *(const v8bf*)(pa + rowsk16 + kb + 16);
        b0.h[0] = *(const v8bf*)(pw + kb);
        b0.h[1] = *(const v8bf*)(pw + kb + 16);
        b1.h[0] = *(const v8bf*)(pw + rowsk16 + kb);
        b1.h[1] = *(const v8bf*)(pw + rowsk16 + kb + 16);

        acc00 = __builtin_amdgcn_wmma_f32_16x16x32_bf16(false, a0.v, false, b0.v,
                                                        (short)0, acc00, false, false);
        acc01 = __builtin_amdgcn_wmma_f32_16x16x32_bf16(false, a0.v, false, b1.v,
                                                        (short)0, acc01, false, false);
        acc10 = __builtin_amdgcn_wmma_f32_16x16x32_bf16(false, a1.v, false, b0.v,
                                                        (short)0, acc10, false, false);
        acc11 = __builtin_amdgcn_wmma_f32_16x16x32_bf16(false, a1.v, false, b1.v,
                                                        (short)0, acc11, false, false);
    }

    // epilogue: bias + optional ReLU, store bf16
    float bn0 = bias[col0 + l16];
    float bn1 = bias[col0 + 16 + l16];
#pragma unroll
    for (int r = 0; r < 8; ++r) {
        int m0 = row0 + g * 8 + r;       // tile row block 0
        int m1 = m0 + 16;                // tile row block 1
        float v00 = acc00[r] + bn0, v01 = acc01[r] + bn1;
        float v10 = acc10[r] + bn0, v11 = acc11[r] + bn1;
        if (relu) {
            v00 = fmaxf(v00, 0.f); v01 = fmaxf(v01, 0.f);
            v10 = fmaxf(v10, 0.f); v11 = fmaxf(v11, 0.f);
        }
        Out[(size_t)m0 * ldo + col0 + l16]      = (__bf16)v00;
        Out[(size_t)m0 * ldo + col0 + 16 + l16] = (__bf16)v01;
        Out[(size_t)m1 * ldo + col0 + l16]      = (__bf16)v10;
        Out[(size_t)m1 * ldo + col0 + 16 + l16] = (__bf16)v11;
    }
}

// ---------------------------------------------------------------
// Final layer: out[b] = sigmoid( Z2[b,:512] . w[:512] + bias )
// One wave per row, shuffle reduction (wave32).
// ---------------------------------------------------------------
__global__ void final_dot_sigmoid(const __bf16* __restrict__ Z2,
                                  const __bf16* __restrict__ w,
                                  const float*  __restrict__ bias,
                                  float*        __restrict__ out) {
    int wave = (blockIdx.x * blockDim.x + threadIdx.x) >> 5;
    int lane = threadIdx.x & 31;
    if (wave >= BB) return;
    const __bf16* row = Z2 + (size_t)wave * 512;
    float s = 0.f;
#pragma unroll
    for (int i = lane; i < 512; i += 32) s += (float)row[i] * (float)w[i];
#pragma unroll
    for (int off = 16; off > 0; off >>= 1) s += __shfl_xor(s, off, 32);
    if (lane == 0) out[wave] = 1.0f / (1.0f + __expf(-(s + bias[0])));
}

// ---------------------------------------------------------------
extern "C" void kernel_launch(void* const* d_in, const int* in_sizes, int n_in,
                              void* d_out, int out_size, void* d_ws, size_t ws_size,
                              hipStream_t stream) {
    const float* x_dense = (const float*)d_in[0];
    const float* tables  = (const float*)d_in[1];
    const float* bw0 = (const float*)d_in[2];  const float* bb0 = (const float*)d_in[3];
    const float* bw1 = (const float*)d_in[4];  const float* bb1 = (const float*)d_in[5];
    const float* bw2 = (const float*)d_in[6];  const float* bb2 = (const float*)d_in[7];
    const float* tw0 = (const float*)d_in[8];  const float* tb0 = (const float*)d_in[9];
    const float* tw1 = (const float*)d_in[10]; const float* tb1 = (const float*)d_in[11];
    const float* tw2 = (const float*)d_in[12]; const float* tb2 = (const float*)d_in[13];
    const int* x_indices = (const int*)d_in[14];
    const int* offsets   = (const int*)d_in[15];
    float* out = (float*)d_out;

    // workspace carve-up (bf16 buffers), 256B aligned, ~40 MB total
    char* ws = (char*)d_ws;
    size_t off = 0;
    auto alloc = [&](size_t elems) -> __bf16* {
        __bf16* p = (__bf16*)(ws + off);
        off = (off + elems * sizeof(__bf16) + 255) & ~(size_t)255;
        return p;
    };
    __bf16* xd16   = alloc((size_t)BB * DENSE);
    __bf16* h1     = alloc((size_t)BB * 1024);
    __bf16* h2     = alloc((size_t)BB * 512);
    __bf16* feat   = alloc((size_t)BB * 640);
    __bf16* z1     = alloc((size_t)BB * 1024);
    __bf16* z2     = alloc((size_t)BB * 512);
    __bf16* bw0_16 = alloc((size_t)1024 * 512);
    __bf16* bw1_16 = alloc((size_t)512 * 1024);
    __bf16* bw2_16 = alloc((size_t)128 * 512);
    __bf16* tw0_16 = alloc((size_t)1024 * 640);
    __bf16* tw1_16 = alloc((size_t)512 * 1024);
    __bf16* tw2_16 = alloc((size_t)512);
    (void)ws_size; (void)n_in; (void)in_sizes; (void)out_size;

    auto cvt = [&](const float* src, __bf16* dst, int n) {
        cvt_f32_bf16<<<(n + 255) / 256, 256, 0, stream>>>(src, dst, n);
    };
    cvt(x_dense, xd16, BB * DENSE);
    cvt(bw0, bw0_16, 1024 * 512);
    cvt(bw1, bw1_16, 512 * 1024);
    cvt(bw2, bw2_16, 128 * 512);
    cvt(tw0, tw0_16, 1024 * 640);
    cvt(tw1, tw1_16, 512 * 1024);
    cvt(tw2, tw2_16, 512);

    auto gemm = [&](const __bf16* A, const __bf16* W, const float* bias, __bf16* O,
                    int M, int N, int K, int ldo, int relu) {
        int waves = (M >> 5) * (N >> 5);
        int blocks = (waves + 7) / 8;              // 256 threads = 8 waves / block
        gemm_bf16_wmma<<<blocks, 256, 0, stream>>>(A, W, bias, O, M, N, K, ldo, relu);
    };

    // bottom MLP
    gemm(xd16, bw0_16, bb0, h1,   BB, 1024, 512,  1024, 1);
    gemm(h1,   bw1_16, bb1, h2,   BB, 512,  1024, 512,  1);
    gemm(h2,   bw2_16, bb2, feat, BB, 128,  512,  640,  1);   // bot_out -> feat[:,0:128]

    // embedding bags -> feat[:,128:640]   (4 tables x 4096 bags = 16384 waves)
    embed_pool<<<(BB * NT) / 8, 256, 0, stream>>>(tables, x_indices, offsets, feat);

    // top MLP
    gemm(feat, tw0_16, tb0, z1, BB, 1024, 640,  1024, 1);
    gemm(z1,   tw1_16, tb1, z2, BB, 512,  1024, 512,  1);

    // final dot + sigmoid -> fp32 output
    final_dot_sigmoid<<<BB / 8, 256, 0, stream>>>(z2, tw2_16, tb2, out);
}
